// ConditioningEncoder_11201274708433
// MI455X (gfx1250) — matrix operand order repaired and checked by reading, not compiled, not run on hardware
//
#include <hip/hip_runtime.h>
#include <hip/hip_bf16.h>

typedef __attribute__((ext_vector_type(16))) _Float16 v16h;
typedef __attribute__((ext_vector_type(4)))  _Float16 v4h;
typedef __attribute__((ext_vector_type(8)))  float    v8f;

#define BATCH   8
#define SPEC    80
#define NEMBD   1024
#define LSEQ    1024
#define NHEAD   16
#define DH      64
#define NLAYER  6
#define GROUPS  32

// ---------------------------------------------------------------------------
// WMMA helpers (contiguous k order k = 16h + s, wave32)
// ---------------------------------------------------------------------------
__device__ __forceinline__ v8f wmma_f16(v16h a, v16h b, v8f c) {
  return __builtin_amdgcn_wmma_f32_16x16x32_f16(false, a, false, b, (short)0, c,
                                                false, false);
}

// A fragment (16x32, M rows): lane m holds row m.
// lanes 0-15: e[0..15]=K0..15 ; lanes 16-31: K16..31 (contiguous)
__device__ __forceinline__ v16h load_a_h(const _Float16* M, int ld, int row, int k0, int lane) {
  const _Float16* rp = M + (size_t)row * ld + k0 + ((lane & 16) ? 16 : 0);
  v16h a;
#pragma unroll
  for (int i = 0; i < 8; ++i) { a[i] = rp[i]; a[8 + i] = rp[8 + i]; }
  return a;
}
// B fragment (32x16, N=lane&15), source stored transposed: Bt[n][k] row-major.
// lanes 0-15 K0..15, lanes 16-31 K16..31 (contiguous 16 halves)
__device__ __forceinline__ v16h load_b_h(const _Float16* Bt, int ld, int nrow, int k0, int lane) {
  const _Float16* rp = Bt + (size_t)nrow * ld + k0 + ((lane & 16) ? 16 : 0);
  v16h b;
#pragma unroll
  for (int i = 0; i < 16; ++i) b[i] = rp[i];
  return b;
}

__device__ __forceinline__ float rmax16(float v) {
#pragma unroll
  for (int off = 1; off < 16; off <<= 1) v = fmaxf(v, __shfl_xor(v, off, 32));
  return v;
}
__device__ __forceinline__ float rsum16(float v) {
#pragma unroll
  for (int off = 1; off < 16; off <<= 1) v += __shfl_xor(v, off, 32);
  return v;
}

// ---------------------------------------------------------------------------
// weight conversion f32 -> f16 (vectorized 4-wide), run once per call
// ---------------------------------------------------------------------------
__global__ __launch_bounds__(256) void k_cvt(const float* __restrict__ src,
                                             _Float16* __restrict__ dst, int n4) {
  int i = blockIdx.x * 256 + threadIdx.x;
  if (i < n4) {
    float4 v = ((const float4*)src)[i];
    v4h h;
    h[0] = (_Float16)v.x; h[1] = (_Float16)v.y;
    h[2] = (_Float16)v.z; h[3] = (_Float16)v.w;
    ((v4h*)dst)[i] = h;
  }
}

// ---------------------------------------------------------------------------
// init: x[b,o,l] = sum_c init_w[o,c]*speech[b,c,l] + init_b[o]   (K=80, VALU fp32)
// ---------------------------------------------------------------------------
__global__ __launch_bounds__(256) void k_init(const float* __restrict__ speech,
                                              const float* __restrict__ w,
                                              const float* __restrict__ bia,
                                              float* __restrict__ x) {
  int l = blockIdx.x * 256 + threadIdx.x;
  int o = blockIdx.y, b = blockIdx.z;
  const float* wr = w + (size_t)o * SPEC;
  const float* sp = speech + (size_t)b * SPEC * LSEQ + l;
  float acc = bia[o];
#pragma unroll 8
  for (int c = 0; c < SPEC; ++c) acc += wr[c] * sp[(size_t)c * LSEQ];
  x[((size_t)b * NEMBD + o) * LSEQ + l] = acc;
}

// ---------------------------------------------------------------------------
// group-norm: x (B,C,L) f32 -> hT (B,L,C) f16 (transposed via LDS tiles)
// ---------------------------------------------------------------------------
__global__ __launch_bounds__(256) void k_gn(const float* __restrict__ x,
                                            const float* __restrict__ gw,
                                            const float* __restrict__ gb,
                                            _Float16* __restrict__ hT) {
  int g = blockIdx.x, b = blockIdx.y, tid = threadIdx.x;
  const float* base = x + ((size_t)b * NEMBD + g * 32) * LSEQ;  // 32 contiguous rows
  float s = 0.f, s2 = 0.f;
  for (int i = tid; i < 32 * LSEQ; i += 256) { float v = base[i]; s += v; s2 += v * v; }
  __shared__ float rs[256], rq[256];
  rs[tid] = s; rq[tid] = s2;
  __syncthreads();
  for (int st = 128; st > 0; st >>= 1) {
    if (tid < st) { rs[tid] += rs[tid + st]; rq[tid] += rq[tid + st]; }
    __syncthreads();
  }
  float mean = rs[0] * (1.f / (32.f * LSEQ));
  float var  = rq[0] * (1.f / (32.f * LSEQ)) - mean * mean;
  float rstd = rsqrtf(fmaxf(var, 0.f) + 1e-5f);

  int cc = tid & 31, lr = tid >> 5;
  int lc = tid & 31, cr = tid >> 5;
  float wsc = gw[g * 32 + cc] * rstd;
  float bsc = gb[g * 32 + cc];
  __shared__ float tile[32][33];
  for (int l0 = 0; l0 < LSEQ; l0 += 32) {
    __syncthreads();
#pragma unroll
    for (int c = cr; c < 32; c += 8) tile[c][lc] = base[(size_t)c * LSEQ + l0 + lc];
    __syncthreads();
#pragma unroll
    for (int l = lr; l < 32; l += 8) {
      float v = (tile[cc][l] - mean) * wsc + bsc;
      hT[((size_t)b * LSEQ + l0 + l) * NEMBD + g * 32 + cc] = (_Float16)v;
    }
  }
}

// ---------------------------------------------------------------------------
// qkv GEMM: D = qkv_w16 (3072x1024 f16) @ h, h transposed (L,C) f16.
// Block: 8 waves = 2(o) x 4(l); wave tile 64(o) x 32(l); 8 WMMA per k-step.
// Epilogue scatters to qT (B,H,L,DH), kT (B,H,L,DH), v (B,H,DH,L), f16.
// ---------------------------------------------------------------------------
__global__ __launch_bounds__(256) void k_qkv(const _Float16* __restrict__ hT,
                                             const _Float16* __restrict__ W16,
                                             const float* __restrict__ bias,
                                             _Float16* __restrict__ qT,
                                             _Float16* __restrict__ kT,
                                             _Float16* __restrict__ vM) {
  int tid = threadIdx.x, lane = tid & 31, wv = tid >> 5;
  int wo = wv >> 2, wl = wv & 3;
  int ob = blockIdx.x * 128 + wo * 64;
  int l0 = blockIdx.y * 128 + wl * 32;
  int b  = blockIdx.z;
  int n  = lane & 15;

  v8f acc[4][2];
#pragma unroll
  for (int m = 0; m < 4; ++m)
#pragma unroll
    for (int s = 0; s < 2; ++s)
#pragma unroll
      for (int i = 0; i < 8; ++i) acc[m][s][i] = 0.f;

  const _Float16* hb = hT + (size_t)b * LSEQ * NEMBD;
  for (int k0 = 0; k0 < NEMBD; k0 += 32) {
    v16h bf0 = load_b_h(hb, NEMBD, l0 + n, k0, lane);
    v16h bf1 = load_b_h(hb, NEMBD, l0 + 16 + n, k0, lane);
#pragma unroll
    for (int m = 0; m < 4; ++m) {
      v16h a = load_a_h(W16, NEMBD, ob + m * 16 + n, k0, lane);
      acc[m][0] = wmma_f16(a, bf0, acc[m][0]);
      acc[m][1] = wmma_f16(a, bf1, acc[m][1]);
    }
  }

  int hoff = (lane & 16) ? 8 : 0;
#pragma unroll
  for (int m = 0; m < 4; ++m) {
#pragma unroll
    for (int s = 0; s < 2; ++s) {
      int lcol = l0 + s * 16 + n;
#pragma unroll
      for (int r = 0; r < 8; ++r) {
        int og = ob + m * 16 + hoff + r;
        _Float16 hv = (_Float16)(acc[m][s][r] + bias[og]);
        int hh = og / 192, sec = og % 192;   // (H, 3*dh) split of the 3072 rows
        if (sec < DH)
          qT[(((size_t)b * NHEAD + hh) * LSEQ + lcol) * DH + sec] = hv;
        else if (sec < 2 * DH)
          kT[(((size_t)b * NHEAD + hh) * LSEQ + lcol) * DH + (sec - DH)] = hv;
        else
          vM[(((size_t)b * NHEAD + hh) * DH + (sec - 2 * DH)) * LSEQ + lcol] = hv;
      }
    }
  }
}

// ---------------------------------------------------------------------------
// flash attention: per wave one 16-row t tile; s-steps of 32.
// scores = (qT tile) x (kT tile), online softmax, P via LDS -> B frag,
// acc (c x t) = v(A) x P(B).  Output written transposed (B,L,C) f16.
// ---------------------------------------------------------------------------
__global__ __launch_bounds__(256) void k_attn(const _Float16* __restrict__ qT,
                                              const _Float16* __restrict__ kT,
                                              const _Float16* __restrict__ vM,
                                              _Float16* __restrict__ aT) {
  int tid = threadIdx.x, lane = tid & 31, wv = tid >> 5;
  int h = blockIdx.y, b = blockIdx.z;
  int t0 = blockIdx.x * 128 + wv * 16;
  int n    = lane & 15;
  int kb   = (lane & 16) ? 16 : 0;
  int boff = (lane & 16) ? 16 : 0;
  int hoff = (lane & 16) ? 8 : 0;

  const _Float16* qb = qT + ((size_t)(b * NHEAD + h) * LSEQ) * DH;
  const _Float16* kbse = kT + ((size_t)(b * NHEAD + h) * LSEQ) * DH;
  const _Float16* vb = vM + (size_t)(b * NHEAD + h) * DH * LSEQ;

  const _Float16* qrow = qb + (size_t)(t0 + n) * DH;
  v16h aq0, aq1;
#pragma unroll
  for (int i = 0; i < 8; ++i) {
    aq0[i] = qrow[kb + i];       aq0[8 + i] = qrow[8 + kb + i];
    aq1[i] = qrow[32 + kb + i];  aq1[8 + i] = qrow[40 + kb + i];
  }

  v8f acc[4];
#pragma unroll
  for (int ci = 0; ci < 4; ++ci)
#pragma unroll
    for (int i = 0; i < 8; ++i) acc[ci][i] = 0.f;
  float mo[8], ls[8];
#pragma unroll
  for (int r = 0; r < 8; ++r) { mo[r] = -3.0e38f; ls[r] = 0.f; }

  __shared__ _Float16 Pl[8][16][32];
  __shared__ float    Cl[8][16];

  for (int s0 = 0; s0 < LSEQ; s0 += 32) {
    const _Float16* kr0 = kbse + (size_t)(s0 + n) * DH;
    const _Float16* kr1 = kr0 + 16 * DH;
    v16h b00, b01, b10, b11;
#pragma unroll
    for (int i = 0; i < 16; ++i) {
      b00[i] = kr0[boff + i];      b01[i] = kr0[32 + boff + i];
      b10[i] = kr1[boff + i];      b11[i] = kr1[32 + boff + i];
    }
    v8f z;
#pragma unroll
    for (int i = 0; i < 8; ++i) z[i] = 0.f;
    v8f sc0 = wmma_f16(aq1, b01, wmma_f16(aq0, b00, z));
    v8f sc1 = wmma_f16(aq1, b11, wmma_f16(aq0, b10, z));

#pragma unroll
    for (int r = 0; r < 8; ++r) {
      float x0 = sc0[r] * 0.125f, x1 = sc1[r] * 0.125f;   // (dh^-1/4)^2 folded in
      float mn = fmaxf(mo[r], rmax16(fmaxf(x0, x1)));
      float corr = __expf(mo[r] - mn);
      float p0 = __expf(x0 - mn), p1 = __expf(x1 - mn);
      ls[r] = ls[r] * corr + rsum16(p0 + p1);
      mo[r] = mn;
      Pl[wv][hoff + r][n]      = (_Float16)p0;
      Pl[wv][hoff + r][16 + n] = (_Float16)p1;
      if (n == r) Cl[wv][hoff + r] = corr;                // one lane/half per row
    }
    float f = Cl[wv][n];                                   // per-t rescale
#pragma unroll
    for (int ci = 0; ci < 4; ++ci)
#pragma unroll
      for (int i = 0; i < 8; ++i) acc[ci][i] *= f;

    v16h bp;                                               // P as B frag (K=s, N=t)
#pragma unroll
    for (int i = 0; i < 16; ++i) bp[i] = Pl[wv][n][boff + i];
#pragma unroll
    for (int ci = 0; ci < 4; ++ci) {
      const _Float16* vr = vb + (size_t)(ci * 16 + n) * LSEQ + s0 + kb;
      v16h av;
#pragma unroll
      for (int i = 0; i < 8; ++i) { av[i] = vr[i]; av[8 + i] = vr[8 + i]; }
      acc[ci] = wmma_f16(av, bp, acc[ci]);
    }
  }

#pragma unroll
  for (int r = 0; r < 8; ++r)
    if (n == r) Cl[wv][hoff + r] = ls[r];
  float linv = 1.f / Cl[wv][n];
  int t = t0 + n;
#pragma unroll
  for (int ci = 0; ci < 4; ++ci)
#pragma unroll
    for (int r = 0; r < 8; ++r) {
      int c = h * DH + ci * 16 + hoff + r;
      aT[((size_t)b * LSEQ + t) * NEMBD + c] = (_Float16)(acc[ci][r] * linv);
    }
}

// ---------------------------------------------------------------------------
// proj GEMM + residual: x[b,o,l] += proj_w16 @ attn(+bias), attn as (L,C) f16
// Block: 8 waves = 2(o) x 4(l); wave tile 64(o) x 32(l)
// ---------------------------------------------------------------------------
__global__ __launch_bounds__(256) void k_proj(const _Float16* __restrict__ aT,
                                              const _Float16* __restrict__ W16,
                                              const float* __restrict__ bias,
                                              float* __restrict__ x) {
  int tid = threadIdx.x, lane = tid & 31, wv = tid >> 5;
  int wo = wv >> 2, wl = wv & 3;
  int ob = blockIdx.x * 128 + wo * 64;
  int l0 = blockIdx.y * 128 + wl * 32;
  int b  = blockIdx.z;
  int n  = lane & 15;

  v8f acc[4][2];
#pragma unroll
  for (int m = 0; m < 4; ++m)
#pragma unroll
    for (int s = 0; s < 2; ++s)
#pragma unroll
      for (int i = 0; i < 8; ++i) acc[m][s][i] = 0.f;

  const _Float16* ab = aT + (size_t)b * LSEQ * NEMBD;
  for (int k0 = 0; k0 < NEMBD; k0 += 32) {
    v16h bf0 = load_b_h(ab, NEMBD, l0 + n, k0, lane);
    v16h bf1 = load_b_h(ab, NEMBD, l0 + 16 + n, k0, lane);
#pragma unroll
    for (int m = 0; m < 4; ++m) {
      v16h a = load_a_h(W16, NEMBD, ob + m * 16 + n, k0, lane);
      acc[m][0] = wmma_f16(a, bf0, acc[m][0]);
      acc[m][1] = wmma_f16(a, bf1, acc[m][1]);
    }
  }
  int hoff = (lane & 16) ? 8 : 0;
#pragma unroll
  for (int m = 0; m < 4; ++m) {
#pragma unroll
    for (int s = 0; s < 2; ++s) {
      int lcol = l0 + s * 16 + n;
#pragma unroll
      for (int r = 0; r < 8; ++r) {
        int og = ob + m * 16 + hoff + r;
        size_t idx = ((size_t)b * NEMBD + og) * LSEQ + lcol;
        x[idx] = x[idx] + acc[m][s][r] + bias[og];
      }
    }
  }
}

__global__ void k_extract(const float* __restrict__ x, float* __restrict__ out) {
  int i = blockIdx.x * 256 + threadIdx.x;
  if (i < BATCH * NEMBD) out[i] = x[(size_t)i * LSEQ];   // x[b,c,0]
}

// ---------------------------------------------------------------------------
extern "C" void kernel_launch(void* const* d_in, const int* in_sizes, int n_in,
                              void* d_out, int out_size, void* d_ws, size_t ws_size,
                              hipStream_t stream) {
  const float* speech = (const float*)d_in[0];
  const float* init_w = (const float*)d_in[1];
  const float* init_b = (const float*)d_in[2];
  const float* gn_w   = (const float*)d_in[3];
  const float* gn_b   = (const float*)d_in[4];
  const float* qkv_w  = (const float*)d_in[5];
  const float* qkv_b  = (const float*)d_in[6];
  const float* proj_w = (const float*)d_in[7];
  const float* proj_b = (const float*)d_in[8];

  char* ws = (char*)d_ws;
  const size_t actF32 = (size_t)BATCH * NEMBD * LSEQ * sizeof(float);     // 32 MB
  const size_t actF16 = (size_t)BATCH * NEMBD * LSEQ * sizeof(_Float16);  // 16 MB
  const size_t qkvWN  = (size_t)NLAYER * 3 * NEMBD * NEMBD;               // elems
  const size_t projWN = (size_t)NLAYER * NEMBD * NEMBD;                   // elems

  size_t off = 0;
  float*    x   = (float*)(ws + off);    off += actF32;
  _Float16* hT  = (_Float16*)(ws + off); off += actF16;  // gn out; re-used as attn out
  _Float16* qT  = (_Float16*)(ws + off); off += actF16;
  _Float16* kT  = (_Float16*)(ws + off); off += actF16;
  _Float16* vM  = (_Float16*)(ws + off); off += actF16;
  _Float16* qW16 = (_Float16*)(ws + off); off += qkvWN * sizeof(_Float16);
  _Float16* pW16 = (_Float16*)(ws + off); off += projWN * sizeof(_Float16);

  // one-shot weight conversion to f16 (amortized; keeps GEMM A-path pure f16)
  k_cvt<<<dim3((int)(qkvWN / 4 + 255) / 256), 256, 0, stream>>>(qkv_w, qW16, (int)(qkvWN / 4));
  k_cvt<<<dim3((int)(projWN / 4 + 255) / 256), 256, 0, stream>>>(proj_w, pW16, (int)(projWN / 4));

  k_init<<<dim3(LSEQ / 256, NEMBD, BATCH), 256, 0, stream>>>(speech, init_w, init_b, x);

  for (int l = 0; l < NLAYER; ++l) {
    k_gn<<<dim3(GROUPS, BATCH), 256, 0, stream>>>(x, gn_w + (size_t)l * NEMBD,
                                                  gn_b + (size_t)l * NEMBD, hT);
    k_qkv<<<dim3(3 * NEMBD / 128, LSEQ / 128, BATCH), 256, 0, stream>>>(
        hT, qW16 + (size_t)l * 3 * NEMBD * NEMBD, qkv_b + (size_t)l * 3 * NEMBD,
        qT, kT, vM);
    k_attn<<<dim3(LSEQ / 128, NHEAD, BATCH), 256, 0, stream>>>(qT, kT, vM, hT);
    k_proj<<<dim3(NEMBD / 128, LSEQ / 128, BATCH), 256, 0, stream>>>(
        hT, pW16 + (size_t)l * NEMBD * NEMBD, proj_b + (size_t)l * NEMBD, x);
  }

  k_extract<<<dim3((BATCH * NEMBD + 255) / 256), 256, 0, stream>>>(x, (float*)d_out);
}
